// MaxKGIN_11768210391439
// MI455X (gfx1250) — matrix operand order repaired and checked by reading, not compiled, not run on hardware
//
#include <hip/hip_runtime.h>
#include <stdint.h>

#define NNODES 50000
#define NEDGES 800000
#define HDIM   128
#define OUTDIM 40
#define NLAYERS 3
#define MAXK   32

typedef __attribute__((ext_vector_type(16))) __bf16 v16bf;
typedef __attribute__((ext_vector_type(8)))  float  v8f;
typedef int v4i __attribute__((vector_size(16)));          // matches builtin's int4 pointee

union ABfrag { v16bf v; unsigned u[8]; };

__device__ __forceinline__ unsigned short f32_to_bf16(float f) {
  unsigned u = __float_as_uint(f);
  unsigned r = u + 0x7FFFu + ((u >> 16) & 1u);   // round-to-nearest-even
  return (unsigned short)(r >> 16);
}

// CDNA5 async HBM->LDS copy, 16B per lane, tracked by ASYNCcnt (no VGPR round trip).
__device__ __forceinline__ void async_g2l_b128(const void* g, void* l) {
#if __has_builtin(__builtin_amdgcn_global_load_async_to_lds_b128)
  __builtin_amdgcn_global_load_async_to_lds_b128(
      (__attribute__((address_space(1))) v4i*)const_cast<void*>(g),
      (__attribute__((address_space(3))) v4i*)l, 0, 0);
#else
  asm volatile("global_load_async_to_lds_b128 %0, %1, off"
               :: "v"((unsigned)(unsigned long long)l),
                  "v"((unsigned long long)g)
               : "memory");
#endif
}

__device__ __forceinline__ void wait_asynccnt0() {
#if __has_builtin(__builtin_amdgcn_s_wait_asynccnt)
  __builtin_amdgcn_s_wait_asynccnt(0);
#else
  asm volatile("s_wait_asynccnt 0x0" ::: "memory");
#endif
}

// Convert+transpose one [128, Mout] f32 weight into bf16 Wt[col][k] (128x128 slot,
// zero-padded columns beyond Mout so padded WMMA B fragments are benign).
__global__ void packW(const float* __restrict__ W, unsigned short* __restrict__ Wt, int mout) {
  int i = blockIdx.x * blockDim.x + threadIdx.x;
  if (i >= 128 * 128) return;
  int k = i & 127, o = i >> 7;
  float v = (o < mout) ? W[k * mout + o] : 0.0f;
  Wt[o * 128 + k] = f32_to_bf16(v);
}

// C = act( A @ W + bias ), K = 128 fixed.
// A source is either:
//   - Abf (bf16 [nrows x 128]): staged via GLOBAL_LOAD_ASYNC_TO_LDS_B128, or
//   - alpha*A0 + A1 (f32): converted to bf16 during staging (fusion path).
// Output: Cb (bf16, mout==128) or C (f32, guarded to mout/ldc).
// Block: 128 threads = 4 waves, tile 64 rows x (CT*16) cols. bf16 WMMA, f32 accum.
template<int CT>
__global__ __launch_bounds__(128)
void gemm_bf16_wmma(const unsigned short* __restrict__ Abf,
                    const float* __restrict__ A0, const float* __restrict__ A1,
                    const float* __restrict__ eps_ptr,
                    const unsigned short* __restrict__ Wt,
                    const float* __restrict__ bias,
                    float* __restrict__ C, unsigned short* __restrict__ Cb,
                    int nrows, int mout, int ldc, int do_relu)
{
  // +8 element pad per row: 272B pitch (= 17*16B) rotates banks across rows so the
  // 16-row fragment gathers hit distinct banks; still 16B-aligned for async b128.
  __shared__ unsigned short sA[64 * 136];
  __shared__ unsigned short sW[CT * 16 * 136];

  const int t = threadIdx.x;
  const int row0 = blockIdx.x * 64;

  // Stage W tile: already bf16-packed, async copy HBM->LDS (16 chunks of 16B per row)
  for (int c = t; c < CT * 16 * 16; c += 128) {
    int o = c >> 4, wi = (c & 15) * 16;
    async_g2l_b128((const char*)Wt + (size_t)o * 256 + wi,
                   (char*)(void*)sW + o * 272 + wi);
  }

  if (Abf) {
    // Pure-GEMM input: bf16 activations, async copy straight into LDS
    for (int c = t; c < 64 * 16; c += 128) {
      int r = c >> 4, wi = (c & 15) * 16;
      int gr = row0 + r;
      if (gr < nrows)
        async_g2l_b128((const char*)Abf + (size_t)gr * 256 + wi,
                       (char*)(void*)sA + r * 272 + wi);
      // rows >= nrows: stale LDS only feeds output rows that are never stored
    }
  } else {
    // Fusion path: A_eff = alpha*A0 + A1, f32 -> bf16 on the fly
    float alpha = 1.0f;
    if (eps_ptr) alpha = 1.0f + eps_ptr[0];   // uniform scalar load
    for (int i = t; i < 64 * 128; i += 128) {
      int r = i >> 7, cc = i & 127;
      int gr = row0 + r;
      float v = 0.0f;
      if (gr < nrows) {
        v = alpha * A0[(size_t)gr * HDIM + cc];
        if (A1) v += A1[(size_t)gr * HDIM + cc];
      }
      sA[r * 136 + cc] = f32_to_bf16(v);
    }
  }
  wait_asynccnt0();
  __syncthreads();

  const int wave = t >> 5;
  const int lane = t & 31;
  const int m  = lane & 15;
  const int hi = lane >> 4;

  // A fragments: ISA 16-bit A 16x32 layout.
  // lane(0-15)=row; VGPR g pairs: g<4 -> K=2g(+1), g>=4 -> K=16+2(g-4)(+1); +8 if lane>=16.
  ABfrag a[4];
  const unsigned short* aRow = &sA[(wave * 16 + m) * 136];
  #pragma unroll
  for (int kc = 0; kc < 4; ++kc) {
    #pragma unroll
    for (int g = 0; g < 8; ++g) {
      int kb = kc * 32 + ((g < 4) ? (2 * g) : (16 + 2 * (g - 4))) + 8 * hi;
      a[kc].u[g] = *(const unsigned*)(aRow + kb);   // 2 contiguous bf16
    }
  }

  v8f zero8 = {0.f, 0.f, 0.f, 0.f, 0.f, 0.f, 0.f, 0.f};
  v8f acc[CT];
  #pragma unroll
  for (int ct = 0; ct < CT; ++ct) acc[ct] = zero8;

  // B fragments: same K order as A (g<4 -> K=2g(+1), g>=4 -> K=16+2(g-4)(+1); +8 if lane>=16), lane&15 = col.
  #pragma unroll
  for (int ct = 0; ct < CT; ++ct) {
    const unsigned short* wCol = &sW[(ct * 16 + m) * 136 + 8 * hi];
    #pragma unroll
    for (int kc = 0; kc < 4; ++kc) {
      ABfrag b;
      #pragma unroll
      for (int g = 0; g < 8; ++g)
        b.u[g] = *(const unsigned*)(wCol + kc * 32 + ((g < 4) ? (2 * g) : (16 + 2 * (g - 4))));
      acc[ct] = __builtin_amdgcn_wmma_f32_16x16x32_bf16(
          false, a[kc].v, false, b.v, (short)0, acc[ct], false, false);
    }
  }

  // Epilogue: C/D layout -> VGPR r is row r (lanes 0-15) / row 8+r (lanes 16-31)
  #pragma unroll
  for (int ct = 0; ct < CT; ++ct) {
    int col = ct * 16 + m;
    float bv = (col < mout) ? bias[col] : 0.0f;
    #pragma unroll
    for (int r = 0; r < 8; ++r) {
      int grow = row0 + wave * 16 + r + 8 * hi;
      float v = acc[ct][r] + bv;
      if (do_relu && v < 0.0f) v = 0.0f;
      if (Cb) {
        if (grow < nrows) Cb[(size_t)grow * HDIM + col] = f32_to_bf16(v);
      } else {
        if (grow < nrows && col < mout) C[(size_t)grow * ldc + col] = v;
      }
    }
  }
}

// Per-row top-32 of 128 values: one wave32 per row, 4 values/lane.
// Radix-select the 32nd-largest on order-preserving uint keys, then ballot-compact
// the >= threshold entries into 32 (f32 value, u8 column) pairs.
__global__ __launch_bounds__(256)
void maxk_select(const float* __restrict__ h, float* __restrict__ vals,
                 unsigned char* __restrict__ idxb)
{
  int row  = blockIdx.x * 8 + (threadIdx.x >> 5);
  int lane = threadIdx.x & 31;
  if (row >= NNODES) return;

  float v[4]; unsigned key[4];
  #pragma unroll
  for (int j = 0; j < 4; ++j) {
    v[j] = h[(size_t)row * HDIM + lane * 4 + j];
    unsigned u = __float_as_uint(v[j]);
    key[j] = u ^ ((u >> 31) ? 0xFFFFFFFFu : 0x80000000u);
  }

  unsigned prefix = 0; int kk = MAXK;
  for (int bit = 31; bit >= 0; --bit) {
    unsigned cand = prefix | (1u << bit);
    unsigned msk  = ~((1u << bit) - 1u);
    int c = 0;
    #pragma unroll
    for (int j = 0; j < 4; ++j) c += ((key[j] & msk) == cand) ? 1 : 0;
    #pragma unroll
    for (int off = 16; off > 0; off >>= 1) c += __shfl_xor(c, off, 32);
    if (c >= kk) prefix = cand; else kk -= c;
  }

  int base = 0;
  #pragma unroll
  for (int j = 0; j < 4; ++j) {
    bool q = key[j] >= prefix;          // matches reference h >= thr (keeps ties)
    unsigned mb = (unsigned)__ballot(q);
    int slot = base + __popc(mb & ((1u << lane) - 1u));
    if (q && slot < MAXK) {
      vals[(size_t)row * MAXK + slot] = v[j];
      idxb[(size_t)row * MAXK + slot] = (unsigned char)(lane * 4 + j);
    }
    base += __popc(mb);
  }
}

__global__ void zerof(float* __restrict__ p, int n) {
  int i = blockIdx.x * blockDim.x + threadIdx.x;
  int s = gridDim.x * blockDim.x;
  for (; i < n; i += s) p[i] = 0.0f;
}

// One wave per edge: 32 lanes scatter the 32 compacted entries of hs[src] into neigh[dst].
// 25.6M f32 atomics; the whole 25.6MB neigh buffer lives in the 192MB L2.
__global__ __launch_bounds__(256)
void scatter_edges(const int* __restrict__ src, const int* __restrict__ dst,
                   const float* __restrict__ vals, const unsigned char* __restrict__ idxb,
                   float* __restrict__ neigh)
{
  int tid  = blockIdx.x * 256 + threadIdx.x;
  int e    = tid >> 5;
  int lane = tid & 31;
  if (e >= NEDGES) return;
  int s = src[e], d = dst[e];
  float v = vals[(size_t)s * MAXK + lane];
  int id  = idxb[(size_t)s * MAXK + lane];
  atomicAdd(neigh + (size_t)d * HDIM + id, v);
}

extern "C" void kernel_launch(void* const* d_in, const int* in_sizes, int n_in,
                              void* d_out, int out_size, void* d_ws, size_t ws_size,
                              hipStream_t stream)
{
  (void)in_sizes; (void)n_in; (void)out_size; (void)ws_size;
  const float* x     = (const float*)d_in[0];
  const int*   src   = (const int*)d_in[1];
  const int*   dst   = (const int*)d_in[2];
  const float* W_in  = (const float*)d_in[3];
  const float* b_in  = (const float*)d_in[4];
  const float* W_lin = (const float*)d_in[5];
  const float* b_lin = (const float*)d_in[6];
  const float* W1    = (const float*)d_in[7];
  const float* b1    = (const float*)d_in[8];
  const float* W2    = (const float*)d_in[9];
  const float* b2    = (const float*)d_in[10];
  const float* eps   = (const float*)d_in[11];
  const float* W_out = (const float*)d_in[12];
  const float* b_out = (const float*)d_in[13];

  // Workspace carve-up (~85 MB)
  char* w = (char*)d_ws;
  unsigned short* Wt = (unsigned short*)w;                 // 11 slots x 128x128 bf16
  size_t off = (size_t)11 * 128 * 128 * 2;
  unsigned short* hbf0 = (unsigned short*)(w + off); off += (size_t)NNODES * HDIM * 2;
  unsigned short* hbf1 = (unsigned short*)(w + off); off += (size_t)NNODES * HDIM * 2;
  float* hf2  = (float*)(w + off); off += (size_t)NNODES * HDIM * 4;
  float* ng   = (float*)(w + off); off += (size_t)NNODES * HDIM * 4;
  float* vals = (float*)(w + off); off += (size_t)NNODES * MAXK * 4;
  unsigned char* idxb = (unsigned char*)(w + off);

  // Pack all weights to transposed bf16 once per launch (graph-capture safe, deterministic)
  packW<<<64, 256, 0, stream>>>(W_in, Wt + 0 * 16384, HDIM);
  for (int i = 0; i < NLAYERS; ++i) {
    packW<<<64, 256, 0, stream>>>(W_lin + (size_t)i * 16384, Wt + (1 + i) * 16384, HDIM);
    packW<<<64, 256, 0, stream>>>(W1    + (size_t)i * 16384, Wt + (4 + i) * 16384, HDIM);
    packW<<<64, 256, 0, stream>>>(W2    + (size_t)i * 16384, Wt + (7 + i) * 16384, HDIM);
  }
  packW<<<64, 256, 0, stream>>>(W_out, Wt + 10 * 16384, OUTDIM);

  dim3 gg((NNODES + 63) / 64), gb(128);

  // h = relu(x @ W_in + b_in) -> bf16 (only ever a GEMM input)
  gemm_bf16_wmma<8><<<gg, gb, 0, stream>>>(nullptr, x, nullptr, nullptr, Wt, b_in,
                                           nullptr, hbf0, NNODES, HDIM, HDIM, 1);

  for (int i = 0; i < NLAYERS; ++i) {
    // h2 = h @ W_lin[i] + b_lin[i] -> f32 (consumed by maxk + eps-fusion)
    gemm_bf16_wmma<8><<<gg, gb, 0, stream>>>(hbf0, nullptr, nullptr, nullptr,
                                             Wt + (1 + i) * 16384, b_lin + i * HDIM,
                                             hf2, nullptr, NNODES, HDIM, HDIM, 0);
    // top-32 sparsify + compact
    maxk_select<<<(NNODES + 7) / 8, 256, 0, stream>>>(hf2, vals, idxb);
    // neigh = 0; scatter-add hs[src] into neigh[dst]
    zerof<<<2048, 256, 0, stream>>>(ng, NNODES * HDIM);
    scatter_edges<<<(NEDGES * 32 + 255) / 256, 256, 0, stream>>>(src, dst, vals, idxb, ng);
    // h3 = relu(((1+eps[i])*h2 + neigh) @ W1[i] + b1[i]) -> bf16
    gemm_bf16_wmma<8><<<gg, gb, 0, stream>>>(nullptr, hf2, ng, eps + i,
                                             Wt + (4 + i) * 16384, b1 + i * HDIM,
                                             nullptr, hbf1, NNODES, HDIM, HDIM, 1);
    // h = h3 @ W2[i] + b2[i] -> bf16 (hbf0 is dead after the first GEMM of this layer)
    gemm_bf16_wmma<8><<<gg, gb, 0, stream>>>(hbf1, nullptr, nullptr, nullptr,
                                             Wt + (7 + i) * 16384, b2 + i * HDIM,
                                             nullptr, hbf0, NNODES, HDIM, HDIM, 0);
  }

  // out = h @ W_out + b_out  (columns padded to 64 in Wt, stores guarded to 40)
  gemm_bf16_wmma<4><<<gg, gb, 0, stream>>>(hbf0, nullptr, nullptr, nullptr,
                                           Wt + 10 * 16384, b_out,
                                           (float*)d_out, nullptr, NNODES, OUTDIM, OUTDIM, 0);
}
